// MambaMIL_49538152792228
// MI455X (gfx1250) — compile-verified, matrix operand order checked by reading, not run on hardware
//
#include <hip/hip_runtime.h>
#include <hip/hip_bf16.h>
#include <cstdint>

// ---------------------------------------------------------------------------
// MambaMIL forward, MI455X (gfx1250, wave32).
//  * All GEMMs: bf16 WMMA (v_wmma_f32_16x16x32_bf16), f32 accumulate,
//    double-buffered LDS staging with native f32->bf16 converts (v_cvt_pk),
//    fused bias/ReLU/tanh/softplus/residual epilogues.
//  * Selective scan: fused dA/dBx computation (avoids 2x256MB of traffic),
//    thread-per-(channel,state), shfl reduction, fused SiLU(z) gating.
// ---------------------------------------------------------------------------

typedef __attribute__((ext_vector_type(16))) __bf16 v16bf;
typedef __attribute__((ext_vector_type(8)))  float  v8f;

// Native converts: backend emits v_cvt_pk_bf16_f32-class ops (RTNE).
static __device__ __forceinline__ uint32_t pk2bf(float x, float y) {
  union { __bf16 b[2]; uint32_t u; } p;
  p.b[0] = (__bf16)x;
  p.b[1] = (__bf16)y;
  return p.u;
}
static __device__ __forceinline__ float silu(float v) {
  return v / (1.0f + __expf(-v));
}

enum { ACT_NONE = 0, ACT_RELU = 1, ACT_TANH = 2, ACT_SOFTPLUS = 3 };

// ---------------------------------------------------------------------------
// Generic fused GEMM: C[m,n] = epilogue( sum_k A[m,k] * W[n,k] + bias[n] )
// Block tile 128x64x32, 8 waves, each wave: 2x2 WMMA 16x16 tiles (32x32).
// Double-buffered LDS (one s_barrier per K step; next tile's global loads
// overlap current tile's WMMAs). Requires M%128==0, N%64==0, K%32==0
// (true for every call below).
// ---------------------------------------------------------------------------
template <int ACT, bool HAS_BIAS, bool RESID, bool CONCAT>
__global__ __launch_bounds__(256) void gemm_wmma_k(
    const float* __restrict__ A0, const float* __restrict__ A1, int lda,
    int ksplit, const float* __restrict__ W, int ldw,
    const float* __restrict__ bias, const float* __restrict__ resid,
    float* __restrict__ C, int ldc, int K) {
  constexpr int BM = 128, BN = 64, BK = 32;
  constexpr int LT = 40;  // ushort row stride (32 + 8 pad) -> 80B, 16B aligned
  __shared__ uint16_t sA[2][BM * LT];  // 2 x 10240 B
  __shared__ uint16_t sB[2][BN * LT];  // 2 x  5120 B

  const int tid  = threadIdx.x;
  const int lane = tid & 31;
  const int wave = tid >> 5;
  const int m0 = blockIdx.y * BM;
  const int n0 = blockIdx.x * BN;
  const int wm = (wave & 3) * 32;   // 4 waves along M
  const int wn = (wave >> 2) * 32;  // 2 waves along N

  v8f zero = {};
  v8f acc[2][2];
  acc[0][0] = zero; acc[0][1] = zero; acc[1][0] = zero; acc[1][1] = zero;

  // ISA 7.12.2 fragment addressing (16-bit A 16x32 / B 32x16):
  // lane L: row/col = L%16 ; elements 0..7 hold K = k0f..k0f+7,
  // elements 8..15 hold K = k0f+16..k0f+23, with k0f = (L>=16)*8.
  const int fr_row = lane & 15;
  const int fr_k0  = (lane >> 4) * 8;

  union Frag { v16bf v; uint4 q[2]; };

  // Stage one 128x32 (A) + 64x32 (W) f32 tile into bf16 LDS buffer `buf`.
  auto stage = [&](int k0, int buf) {
#pragma unroll
    for (int it = 0; it < 4; ++it) {
      int idx = (it * 256 + tid) * 4;
      int r = idx >> 5, c = idx & 31;
      float4 v;
      if (CONCAT) {
        int gk = k0 + c;
        const float* src = (gk < ksplit)
            ? (A0 + (size_t)(m0 + r) * ksplit + gk)
            : (A1 + (size_t)(m0 + r) * ksplit + (gk - ksplit));
        v = *reinterpret_cast<const float4*>(src);
      } else {
        v = *reinterpret_cast<const float4*>(A0 + (size_t)(m0 + r) * lda + k0 + c);
      }
      *reinterpret_cast<uint2*>(&sA[buf][r * LT + c]) =
          make_uint2(pk2bf(v.x, v.y), pk2bf(v.z, v.w));
    }
#pragma unroll
    for (int it = 0; it < 2; ++it) {
      int idx = (it * 256 + tid) * 4;
      int r = idx >> 5, c = idx & 31;
      float4 v = *reinterpret_cast<const float4*>(W + (size_t)(n0 + r) * ldw + k0 + c);
      *reinterpret_cast<uint2*>(&sB[buf][r * LT + c]) =
          make_uint2(pk2bf(v.x, v.y), pk2bf(v.z, v.w));
    }
  };

  stage(0, 0);
  int buf = 0;
  for (int k0 = 0; k0 < K; k0 += BK, buf ^= 1) {
    __syncthreads();                     // staging of `buf` complete
    if (k0 + BK < K) stage(k0 + BK, buf ^ 1);  // overlap with WMMAs below

    Frag a[2], b[2];
#pragma unroll
    for (int i = 0; i < 2; ++i) {
      const uint16_t* base = &sA[buf][(wm + i * 16 + fr_row) * LT];
      a[i].q[0] = *reinterpret_cast<const uint4*>(base + fr_k0);
      a[i].q[1] = *reinterpret_cast<const uint4*>(base + fr_k0 + 16);
    }
#pragma unroll
    for (int j = 0; j < 2; ++j) {
      const uint16_t* base = &sB[buf][(wn + j * 16 + fr_row) * LT];
      b[j].q[0] = *reinterpret_cast<const uint4*>(base + fr_k0);
      b[j].q[1] = *reinterpret_cast<const uint4*>(base + fr_k0 + 16);
    }
#pragma unroll
    for (int i = 0; i < 2; ++i)
#pragma unroll
      for (int j = 0; j < 2; ++j)
        acc[i][j] = __builtin_amdgcn_wmma_f32_16x16x32_bf16(
            false, a[i].v, false, b[j].v, (short)0, acc[i][j], false, false);
  }

  // ---- epilogue: bias + activation (+ residual) ----
  const int rowAdd = (lane >> 4) * 8;  // C/D: VGPR r -> M = r + (lane>=16)*8
#pragma unroll
  for (int i = 0; i < 2; ++i)
#pragma unroll
    for (int j = 0; j < 2; ++j) {
      int ncol = n0 + wn + j * 16 + (lane & 15);
      float bv = HAS_BIAS ? bias[ncol] : 0.0f;
#pragma unroll
      for (int r = 0; r < 8; ++r) {
        int row = m0 + wm + i * 16 + r + rowAdd;
        float v = acc[i][j][r] + bv;
        if (ACT == ACT_RELU)      v = fmaxf(v, 0.0f);
        else if (ACT == ACT_TANH) v = tanhf(v);
        else if (ACT == ACT_SOFTPLUS) v = (v > 20.0f) ? v : log1pf(__expf(v));
        if (RESID) v += resid[(size_t)row * ldc + ncol];
        C[(size_t)row * ldc + ncol] = v;
      }
    }
}

// ---------------------------------------------------------------------------
// LayerNorm over D=512, one wave per row (8 rows / 256-thread block).
// ---------------------------------------------------------------------------
__global__ __launch_bounds__(256) void layernorm_k(
    const float* __restrict__ x, const float* __restrict__ w,
    const float* __restrict__ b, float* __restrict__ y, int Dm) {
  int wave = threadIdx.x >> 5, lane = threadIdx.x & 31;
  int row = blockIdx.x * 8 + wave;
  const float* xr = x + (size_t)row * Dm;
  float s = 0.0f;
  for (int c = lane; c < Dm; c += 32) s += xr[c];
  for (int m = 16; m >= 1; m >>= 1) s += __shfl_xor(s, m, 32);
  float mean = s / (float)Dm;
  float v = 0.0f;
  for (int c = lane; c < Dm; c += 32) { float d = xr[c] - mean; v += d * d; }
  for (int m = 16; m >= 1; m >>= 1) v += __shfl_xor(v, m, 32);
  float inv = rsqrtf(v / (float)Dm + 1e-5f);
  float* yr = y + (size_t)row * Dm;
  for (int c = lane; c < Dm; c += 32) yr[c] = (xr[c] - mean) * inv * w[c] + b[c];
}

// ---------------------------------------------------------------------------
// Depthwise causal conv (4 taps) + bias + SiLU. x is the first DI=1024 cols
// of xz (row stride 2048). out: (L,1024).
// ---------------------------------------------------------------------------
__global__ __launch_bounds__(256) void conv_silu_k(
    const float* __restrict__ xz, const float* __restrict__ cw,
    const float* __restrict__ cb, float* __restrict__ out) {
  int idx = blockIdx.x * 256 + threadIdx.x;  // l*1024 + di
  int l = idx >> 10, di = idx & 1023;
  float acc = cb[di];
#pragma unroll
  for (int k = 0; k < 4; ++k) {
    int ls = l + k - 3;
    float xv = (ls >= 0) ? xz[(size_t)ls * 2048 + di] : 0.0f;
    acc = fmaf(xv, cw[di * 4 + k], acc);
  }
  out[idx] = silu(acc);
}

// ---------------------------------------------------------------------------
// Selective scan, fully fused: dA=exp(dt*A), dBx=dt*B*x built in registers.
// Thread = (channel, state): 64 blocks x 256 thr = 1024 ch x 16 states.
// 64-step LDS tiles for B/C/dt/x/z; shfl_xor reduction over 16 states;
// write y = (sum_s h*C + D*x) * silu(z).
// ---------------------------------------------------------------------------
__global__ __launch_bounds__(256) void scan_k(
    const float* __restrict__ xdbl, const float* __restrict__ dtb,
    const float* __restrict__ xconv, const float* __restrict__ xz,
    const float* __restrict__ A_log, const float* __restrict__ Dp,
    float* __restrict__ yout, int L) {
  constexpr int TL = 64;
  __shared__ float sB[TL * 16], sC[TL * 16], sdt[TL * 16], sx[TL * 16], sz[TL * 16];
  const int tid = threadIdx.x;
  const int s = tid & 15, ch = tid >> 4;
  const int di0 = blockIdx.x * 16;
  const int di = di0 + ch;
  const float Aval = -__expf(A_log[di * 16 + s]);
  const float dpv  = Dp[di];
  float hst = 0.0f;

  for (int t0 = 0; t0 < L; t0 += TL) {
#pragma unroll
    for (int it = 0; it < 4; ++it) {
      int idx = it * 256 + tid;
      int st = idx >> 4, j = idx & 15;
      size_t l = (size_t)(t0 + st);
      sB[idx]  = xdbl[l * 64 + 32 + j];
      sC[idx]  = xdbl[l * 64 + 48 + j];
      sdt[idx] = dtb[l * 1024 + di0 + j];
      sx[idx]  = xconv[l * 1024 + di0 + j];
      sz[idx]  = xz[l * 2048 + 1024 + di0 + j];
    }
    __syncthreads();
    if (t0 + TL < L) {  // global_prefetch_b8 for the next tile
      __builtin_prefetch(&dtb[(size_t)(t0 + TL) * 1024 + di], 0, 1);
      __builtin_prefetch(&xconv[(size_t)(t0 + TL) * 1024 + di], 0, 1);
    }
    for (int t = 0; t < TL; ++t) {
      float dtv = sdt[t * 16 + ch];
      float xv  = sx[t * 16 + ch];
      float dA  = __expf(dtv * Aval);                 // dt>=0, A<0 -> (0,1]
      hst = fmaf(dA, hst, dtv * sB[t * 16 + s] * xv); // h = dA*h + dt*B*x
      float yv = hst * sC[t * 16 + s];
      yv += __shfl_xor(yv, 1, 32);
      yv += __shfl_xor(yv, 2, 32);
      yv += __shfl_xor(yv, 4, 32);
      yv += __shfl_xor(yv, 8, 32);
      if (s == 0) {
        float zv = sz[t * 16 + ch];
        yout[(size_t)(t0 + t) * 1024 + di] = (yv + dpv * xv) * silu(zv);
      }
    }
    __syncthreads();
  }
}

// ---------------------------------------------------------------------------
// scores[l] = dot(t[l,:128], w2) + b2, one wave per row.
// ---------------------------------------------------------------------------
__global__ __launch_bounds__(256) void attn_score_k(
    const float* __restrict__ t, const float* __restrict__ w2,
    const float* __restrict__ b2, float* __restrict__ sc) {
  int wave = threadIdx.x >> 5, lane = threadIdx.x & 31;
  int row = blockIdx.x * 8 + wave;
  const float* tr = t + (size_t)row * 128;
  float s = 0.0f;
  for (int c = lane; c < 128; c += 32) s += tr[c] * w2[c];
  for (int m = 16; m >= 1; m >>= 1) s += __shfl_xor(s, m, 32);
  if (lane == 0) sc[row] = s + b2[0];
}

// ---------------------------------------------------------------------------
// Softmax over 4096 scores + weighted pooling: pooled = softmax(sc) @ hf.
// Single block, 512 threads; probs cached in LDS (16 KB of 320 KB WGP LDS).
// ---------------------------------------------------------------------------
__global__ __launch_bounds__(512) void softmax_pool_k(
    const float* __restrict__ sc, const float* __restrict__ hf,
    float* __restrict__ pooled) {
  __shared__ float sm[4096];
  __shared__ float red[16];
  __shared__ float gvals[2];
  int tid = threadIdx.x, lane = tid & 31, wave = tid >> 5;

  float lmax = -1e30f;
  for (int l = tid; l < 4096; l += 512) { float v = sc[l]; sm[l] = v; lmax = fmaxf(lmax, v); }
  for (int m = 16; m >= 1; m >>= 1) lmax = fmaxf(lmax, __shfl_xor(lmax, m, 32));
  if (lane == 0) red[wave] = lmax;
  __syncthreads();
  if (wave == 0) {
    float v = (lane < 16) ? red[lane] : -1e30f;
    for (int m = 16; m >= 1; m >>= 1) v = fmaxf(v, __shfl_xor(v, m, 32));
    if (lane == 0) gvals[0] = v;
  }
  __syncthreads();
  float gmax = gvals[0];

  float lsum = 0.0f;
  for (int l = tid; l < 4096; l += 512) { float e = __expf(sm[l] - gmax); sm[l] = e; lsum += e; }
  for (int m = 16; m >= 1; m >>= 1) lsum += __shfl_xor(lsum, m, 32);
  __syncthreads();
  if (lane == 0) red[wave] = lsum;
  __syncthreads();
  if (wave == 0) {
    float v = (lane < 16) ? red[lane] : 0.0f;
    for (int m = 16; m >= 1; m >>= 1) v += __shfl_xor(v, m, 32);
    if (lane == 0) gvals[1] = v;
  }
  __syncthreads();
  float inv = 1.0f / gvals[1];

  if (tid < 512) {
    float acc = 0.0f;
    for (int l = 0; l < 4096; ++l) acc = fmaf(sm[l], hf[(size_t)l * 512 + tid], acc);
    pooled[tid] = acc * inv;
  }
}

// ---------------------------------------------------------------------------
// Head: logits(4) -> sigmoid/clip -> lce -> softmax -> loss. 9 outputs.
// ---------------------------------------------------------------------------
__global__ void final_head_k(const float* __restrict__ pooled,
                             const float* __restrict__ cls_w,
                             const float* __restrict__ cls_b,
                             const int* __restrict__ label,
                             float* __restrict__ out) {
  int lane = threadIdx.x;
  float p[4] = {0.0f, 0.0f, 0.0f, 0.0f};
  for (int k = lane; k < 512; k += 32) {
    float pv = pooled[k];
#pragma unroll
    for (int c = 0; c < 4; ++c) p[c] = fmaf(pv, cls_w[c * 512 + k], p[c]);
  }
#pragma unroll
  for (int c = 0; c < 4; ++c)
    for (int m = 16; m >= 1; m >>= 1) p[c] += __shfl_xor(p[c], m, 32);
  if (lane == 0) {
    float lce[4];
#pragma unroll
    for (int c = 0; c < 4; ++c) {
      float lg = p[c] + cls_b[c];
      float hz = 1.0f / (1.0f + __expf(-lg));
      float pr = fminf(fmaxf(hz, 1e-6f), 1.0f - 1e-6f);
      lce[c] = logf(pr) - log1pf(-pr);
    }
    float mx = fmaxf(fmaxf(lce[0], lce[1]), fmaxf(lce[2], lce[3]));
    float se = 0.0f;
#pragma unroll
    for (int c = 0; c < 4; ++c) se += __expf(lce[c] - mx);
    float lse = mx + logf(se);
#pragma unroll
    for (int c = 0; c < 4; ++c) { out[c] = lce[c]; out[4 + c] = __expf(lce[c] - lse); }
    int lb = label[0];
    out[8] = -(lce[lb & 3] - lse);
  }
}

// ---------------------------------------------------------------------------
extern "C" void kernel_launch(void* const* d_in, const int* in_sizes, int n_in,
                              void* d_out, int out_size, void* d_ws, size_t ws_size,
                              hipStream_t stream) {
  (void)in_sizes; (void)n_in; (void)out_size; (void)ws_size;
  const float* data_s    = (const float*)d_in[0];
  const float* data_l    = (const float*)d_in[1];
  const int*   label     = (const int*)d_in[3];
  const float* fc1_w     = (const float*)d_in[4];
  const float* fc1_b     = (const float*)d_in[5];
  const float* ln_w      = (const float*)d_in[6];
  const float* ln_b      = (const float*)d_in[7];
  const float* in_proj_w = (const float*)d_in[8];
  const float* conv_w    = (const float*)d_in[9];
  const float* conv_b    = (const float*)d_in[10];
  const float* x_proj_w  = (const float*)d_in[11];
  const float* dt_proj_w = (const float*)d_in[12];
  const float* dt_proj_b = (const float*)d_in[13];
  const float* A_log     = (const float*)d_in[14];
  const float* D_param   = (const float*)d_in[15];
  const float* out_proj_w= (const float*)d_in[16];
  const float* norm_w    = (const float*)d_in[17];
  const float* norm_b    = (const float*)d_in[18];
  const float* attn_w1   = (const float*)d_in[19];
  const float* attn_b1   = (const float*)d_in[20];
  const float* attn_w2   = (const float*)d_in[21];
  const float* attn_b2   = (const float*)d_in[22];
  const float* cls_w     = (const float*)d_in[23];
  const float* cls_b     = (const float*)d_in[24];

  const int N = 4096;
  float* ws = (float*)d_ws;
  size_t off = 0;
  auto alloc = [&](size_t n) { float* p = ws + off; off += (n + 3) & ~(size_t)3; return p; };
  float* h    = alloc((size_t)N * 512);    // running hidden state
  float* hn   = alloc((size_t)N * 512);    // layernorm output
  float* xzb  = alloc((size_t)N * 2048);   // in_proj output [x | z]
  float* xcv  = alloc((size_t)N * 1024);   // conv+silu output
  float* xdl  = alloc((size_t)N * 64);     // x_proj output [dt(32)|B(16)|C(16)]
  float* dtb  = alloc((size_t)N * 1024);   // softplus(dt_proj) — reused as attn t
  float* yb   = alloc((size_t)N * 1024);   // scan output (gated)
  float* sco  = alloc(N);                  // attention scores
  float* pool = alloc(512);                // pooled features

  dim3 blk(256);

  // fc1: relu(concat(data_s,data_l) @ fc1_w.T + b)   M=4096 K=1024 N=512
  gemm_wmma_k<ACT_RELU, true, false, true>
      <<<dim3(512 / 64, N / 128), blk, 0, stream>>>(
          data_s, data_l, 0, 512, fc1_w, 1024, fc1_b, nullptr, h, 512, 1024);

  for (int i = 0; i < 2; ++i) {
    layernorm_k<<<N / 8, blk, 0, stream>>>(h, ln_w + i * 512, ln_b + i * 512, hn, 512);
    // in_proj: M=4096 K=512 N=2048
    gemm_wmma_k<ACT_NONE, false, false, false>
        <<<dim3(2048 / 64, N / 128), blk, 0, stream>>>(
            hn, nullptr, 512, 0, in_proj_w + (size_t)i * 2048 * 512, 512,
            nullptr, nullptr, xzb, 2048, 512);
    conv_silu_k<<<(N * 1024) / 256, blk, 0, stream>>>(
        xzb, conv_w + (size_t)i * 1024 * 4, conv_b + (size_t)i * 1024, xcv);
    // x_proj: M=4096 K=1024 N=64
    gemm_wmma_k<ACT_NONE, false, false, false>
        <<<dim3(1, N / 128), blk, 0, stream>>>(
            xcv, nullptr, 1024, 0, x_proj_w + (size_t)i * 64 * 1024, 1024,
            nullptr, nullptr, xdl, 64, 1024);
    // dt_proj: A = xdl[:, :32] (lda=64), M=4096 K=32 N=1024, softplus+bias
    gemm_wmma_k<ACT_SOFTPLUS, true, false, false>
        <<<dim3(1024 / 64, N / 128), blk, 0, stream>>>(
            xdl, nullptr, 64, 0, dt_proj_w + (size_t)i * 1024 * 32, 32,
            dt_proj_b + (size_t)i * 1024, nullptr, dtb, 1024, 32);
    scan_k<<<64, blk, 0, stream>>>(
        xdl, dtb, xcv, xzb, A_log + (size_t)i * 1024 * 16,
        D_param + (size_t)i * 1024, yb, N);
    // out_proj + residual: M=4096 K=1024 N=512, writes h in place
    gemm_wmma_k<ACT_NONE, false, true, false>
        <<<dim3(512 / 64, N / 128), blk, 0, stream>>>(
            yb, nullptr, 1024, 0, out_proj_w + (size_t)i * 512 * 1024, 1024,
            nullptr, h, h, 512, 1024);
  }

  // final LN -> hn
  layernorm_k<<<N / 8, blk, 0, stream>>>(h, norm_w, norm_b, hn, 512);
  // attn hidden: tanh(hn @ w1.T + b1)  M=4096 K=512 N=128  (reuse dtb)
  gemm_wmma_k<ACT_TANH, true, false, false>
      <<<dim3(128 / 64, N / 128), blk, 0, stream>>>(
          hn, nullptr, 512, 0, attn_w1, 512, attn_b1, nullptr, dtb, 128, 512);
  attn_score_k<<<N / 8, blk, 0, stream>>>(dtb, attn_w2, attn_b2, sco);
  softmax_pool_k<<<1, 512, 0, stream>>>(sco, hn, pool);
  final_head_k<<<1, 32, 0, stream>>>(pool, cls_w, cls_b, label, (float*)d_out);
}
